// LTCModel_9371618639926
// MI455X (gfx1250) — matrix operand order repaired and checked by reading, not compiled, not run on hardware
//
#include <hip/hip_runtime.h>
#include <hip/hip_bf16.h>

// LTC recurrent network, MI455X (gfx1250, wave32).
//   Main scan kernel: VALU/transcendental bound; params live in LDS (320KB/WGP).
//   Readout kernel:   WMMA f32<=f16 16x16x32 matvec (h[512,128] @ fc_w[128]).

#define U        128
#define BATCH    512
#define TSTEPS   256
#define UNFOLDS  6
#define NB       8      // samples per block
#define NBH      4      // samples per thread (two j-groups of 128 threads)
#define THREADS  256
#define LTC_EPS  1e-8f

typedef __attribute__((ext_vector_type(16))) _Float16 v16h;
typedef __attribute__((ext_vector_type(8)))  float    v8f;

__device__ __forceinline__ float sigmoid_fast(float x) {
    // 1 / (1 + exp(-x)) : v_exp_f32 + v_add + v_rcp_f32
    return __builtin_amdgcn_rcpf(1.0f + __expf(-x));
}

// ---------------------------------------------------------------------------
// Scan kernel: one block owns NB samples for the full T loop.
// LDS: s_p[U*U] float4 = (sigma, mu*sigma, w_syn*erev, 0)  -> 256 KB
//      s_v[U][NB] float (state, i-major for float4 broadcast) -> 4 KB
// ---------------------------------------------------------------------------
__global__ __launch_bounds__(THREADS)
void ltc_scan_kernel(const float* __restrict__ x,       // [B, T, 1]
                     const float* __restrict__ gleak,   // [U]
                     const float* __restrict__ vleak,   // [U]
                     const float* __restrict__ cm,      // [U]
                     const float* __restrict__ sigma,   // [U,U] (pre i, post j)
                     const float* __restrict__ mu,      // [U,U]
                     const float* __restrict__ w_syn,   // [U,U]
                     const float* __restrict__ erev,    // [U,U] (+-1)
                     const float* __restrict__ s_sigma, // [1,U]
                     const float* __restrict__ s_mu,    // [1,U]
                     const float* __restrict__ s_w,     // [1,U]
                     const float* __restrict__ s_erev,  // [1,U] (+-1)
                     const float* __restrict__ input_w, // [1]
                     const float* __restrict__ input_b, // [1]
                     const float* __restrict__ output_w,// [U]
                     const float* __restrict__ output_b,// [U]
                     float* __restrict__ h_out)         // [B, U] affine-mapped state
{
    extern __shared__ char smem[];
    float4* s_p = (float4*)smem;                              // [U*U]
    float*  s_v = (float*)(smem + (size_t)U * U * sizeof(float4)); // [U*NB]

    const int tid = threadIdx.x;
    const int j   = tid & (U - 1);   // post-synaptic neuron
    const int g   = tid >> 7;        // which half of the NB samples
    const int b0  = blockIdx.x * NB + g * NBH;

    // Load + precombine parameters into LDS (reused B*T*UNFOLDS times).
    for (int idx = tid; idx < U * U; idx += THREADS) {
        float sg = sigma[idx];
        float m  = mu[idx];
        float w  = w_syn[idx];
        float e  = erev[idx];
        // arg = v*sigma - mu*sigma ; w_syn = fabs(w_syn*erev) since erev=+-1
        s_p[idx] = make_float4(sg, m * sg, w * e, 0.0f);
    }
    for (int idx = tid; idx < U * NB; idx += THREADS) s_v[idx] = 0.0f;

    // Per-j constants.
    const float cmt   = cm[j] * (float)UNFOLDS;
    const float gl    = gleak[j];
    const float glvl  = gl * vleak[j];
    const float dbase = cmt + gl + LTC_EPS;
    const float ssig  = s_sigma[j];
    const float sc    = s_mu[j] * ssig;
    const float swe   = s_w[j] * s_erev[j];
    const float swa   = fabsf(swe);
    const float ow    = output_w[j];
    const float ob    = output_b[j];
    const float inw   = input_w[0];
    const float inb   = input_b[0];

    float vcur[NBH];
#pragma unroll
    for (int lb = 0; lb < NBH; ++lb) vcur[lb] = 0.0f;

    __syncthreads();

    for (int t = 0; t < TSTEPS; ++t) {
        // Sensory conductances (fixed across unfolds). IN_DIM == 1.
        float wnum_s[NBH], wden_s[NBH];
#pragma unroll
        for (int lb = 0; lb < NBH; ++lb) {
            const float* xp = x + (size_t)(b0 + lb) * TSTEPS + t;
            float u  = fmaf(xp[0], inw, inb);
            __builtin_prefetch(xp + 1, 0, 0);          // global_prefetch_b8
            float sa = sigmoid_fast(fmaf(u, ssig, -sc));
            wnum_s[lb] = swe * sa;
            wden_s[lb] = swa * sa;
        }

        for (int k = 0; k < UNFOLDS; ++k) {
            float num[NBH], den[NBH];
#pragma unroll
            for (int lb = 0; lb < NBH; ++lb) { num[lb] = wnum_s[lb]; den[lb] = wden_s[lb]; }

#pragma unroll 4
            for (int i = 0; i < U; ++i) {
                float4 p  = s_p[i * U + j];                       // ds_load_b128, conflict-free
                float4 vv = ((const float4*)s_v)[i * 2 + g];      // ds_load_b128, broadcast
                const float* vvp = (const float*)&vv;
                float wa = fabsf(p.z);
#pragma unroll
                for (int lb = 0; lb < NBH; ++lb) {
                    float s = sigmoid_fast(fmaf(vvp[lb], p.x, -p.y));
                    num[lb] = fmaf(p.z, s, num[lb]);
                    den[lb] = fmaf(wa,  s, den[lb]);
                }
            }

            __syncthreads();   // all reads of s_v done before overwrite
            float4 vnew4;
            float* vn = (float*)&vnew4;
#pragma unroll
            for (int lb = 0; lb < NBH; ++lb) {
                float numer = fmaf(cmt, vcur[lb], glvl) + num[lb];
                float denom = dbase + den[lb];
                float v = numer * __builtin_amdgcn_rcpf(denom);
                vcur[lb] = v;
                vn[lb]   = v;
            }
            ((float4*)s_v)[j * 2 + g] = vnew4;                    // ds_store_b128
            __syncthreads();
        }
    }

    // Affine output mapping, stash final state for the WMMA readout.
#pragma unroll
    for (int lb = 0; lb < NBH; ++lb)
        h_out[(size_t)(b0 + lb) * U + j] = fmaf(vcur[lb], ow, ob);
}

// ---------------------------------------------------------------------------
// Readout: out[b] = h[b,:] . fc_w + fc_b  via v_wmma_f32_16x16x32_f16.
// One wave32 per 16-sample tile; 4 chained WMMAs over K=128; fc_w in column 0.
// All fragment loads are unconditional global_load_b128; the column-0 mask is
// applied branch-free so EXEC stays all-ones through every WMMA.
// ---------------------------------------------------------------------------
__global__ __launch_bounds__(32)
void ltc_readout_wmma(const float* __restrict__ h,     // [B, U]
                      const float* __restrict__ fc_w,  // [1, U]
                      const float* __restrict__ fc_b,  // [1]
                      float* __restrict__ out)         // [B, 1]
{
    const int lane = threadIdx.x;          // 0..31
    const int m    = lane & 15;
    const int ksel = lane >> 4;
    const int row  = blockIdx.x * 16 + m;  // sample index for this lane's A rows

    const float* hrow = h + (size_t)row * U;
    const float  msk  = (m == 0) ? 1.0f : 0.0f;   // v_cndmask, no branch

    v8f c = {};
#pragma unroll
    for (int kc = 0; kc < 4; ++kc) {
        // A fragment: 16x32 f16. Lane L<16: M=L, K in {ksel*8..+7} U {16+ksel*8..+7}.
        const int kA = kc * 32 + ksel * 8;
        float4 a0 = *(const float4*)(hrow + kA);          // K = kA+0..3
        float4 a1 = *(const float4*)(hrow + kA + 4);      // K = kA+4..7
        float4 a2 = *(const float4*)(hrow + kA + 16);     // K = kA+16..19
        float4 a3 = *(const float4*)(hrow + kA + 20);     // K = kA+20..23
        v16h a;
        a[0]  = (_Float16)a0.x; a[1]  = (_Float16)a0.y;
        a[2]  = (_Float16)a0.z; a[3]  = (_Float16)a0.w;
        a[4]  = (_Float16)a1.x; a[5]  = (_Float16)a1.y;
        a[6]  = (_Float16)a1.z; a[7]  = (_Float16)a1.w;
        a[8]  = (_Float16)a2.x; a[9]  = (_Float16)a2.y;
        a[10] = (_Float16)a2.z; a[11] = (_Float16)a2.w;
        a[12] = (_Float16)a3.x; a[13] = (_Float16)a3.y;
        a[14] = (_Float16)a3.z; a[15] = (_Float16)a3.w;

        // B fragment: 32x16 f16, column N=0 = fc_w chunk (masked), rest zero.
        // Lane L: N=L&15, K in {ksel*8..+7} U {16+ksel*8..+7} (same as A).
        const int kB = kc * 32 + ksel * 8;
        float4 w0 = *(const float4*)(fc_w + kB);
        float4 w1 = *(const float4*)(fc_w + kB + 4);
        float4 w2 = *(const float4*)(fc_w + kB + 16);
        float4 w3 = *(const float4*)(fc_w + kB + 20);
        v16h b;
        b[0]  = (_Float16)(w0.x * msk); b[1]  = (_Float16)(w0.y * msk);
        b[2]  = (_Float16)(w0.z * msk); b[3]  = (_Float16)(w0.w * msk);
        b[4]  = (_Float16)(w1.x * msk); b[5]  = (_Float16)(w1.y * msk);
        b[6]  = (_Float16)(w1.z * msk); b[7]  = (_Float16)(w1.w * msk);
        b[8]  = (_Float16)(w2.x * msk); b[9]  = (_Float16)(w2.y * msk);
        b[10] = (_Float16)(w2.z * msk); b[11] = (_Float16)(w2.w * msk);
        b[12] = (_Float16)(w3.x * msk); b[13] = (_Float16)(w3.y * msk);
        b[14] = (_Float16)(w3.z * msk); b[15] = (_Float16)(w3.w * msk);

        // (neg_a, A, neg_b, B, c_mod, C, reuse_a, reuse_b) -- EXEC all-ones here
        c = __builtin_amdgcn_wmma_f32_16x16x32_f16(false, a, false, b,
                                                   (short)0, c, false, false);
    }

    // D column N=0: lane 0 holds M=0..7 in c[0..7], lane 16 holds M=8..15.
    const float bias = fc_b[0];
    if (m == 0) {
        const int mbase = blockIdx.x * 16 + ksel * 8;
#pragma unroll
        for (int r = 0; r < 8; ++r)
            out[mbase + r] = c[r] + bias;
    }
}

// ---------------------------------------------------------------------------
extern "C" void kernel_launch(void* const* d_in, const int* in_sizes, int n_in,
                              void* d_out, int out_size, void* d_ws, size_t ws_size,
                              hipStream_t stream) {
    const float* x        = (const float*)d_in[0];
    const float* gleak    = (const float*)d_in[1];
    const float* vleak    = (const float*)d_in[2];
    const float* cm       = (const float*)d_in[3];
    const float* sigma    = (const float*)d_in[4];
    const float* mu       = (const float*)d_in[5];
    const float* w_syn    = (const float*)d_in[6];
    const float* erev     = (const float*)d_in[7];
    const float* s_sigma  = (const float*)d_in[8];
    const float* s_mu     = (const float*)d_in[9];
    const float* s_w      = (const float*)d_in[10];
    const float* s_erev   = (const float*)d_in[11];
    const float* input_w  = (const float*)d_in[12];
    const float* input_b  = (const float*)d_in[13];
    const float* output_w = (const float*)d_in[14];
    const float* output_b = (const float*)d_in[15];
    const float* fc_w     = (const float*)d_in[16];
    const float* fc_b     = (const float*)d_in[17];

    float* h_ws = (float*)d_ws;  // [BATCH, U] f32 = 256 KB

    const size_t smem = (size_t)U * U * sizeof(float4) + (size_t)U * NB * sizeof(float);

    ltc_scan_kernel<<<BATCH / NB, THREADS, smem, stream>>>(
        x, gleak, vleak, cm, sigma, mu, w_syn, erev,
        s_sigma, s_mu, s_w, s_erev, input_w, input_b, output_w, output_b, h_ws);

    ltc_readout_wmma<<<BATCH / 16, 32, 0, stream>>>(h_ws, fc_w, fc_b, (float*)d_out);
}
